// GraphTransformer_3221225472249
// MI455X (gfx1250) — matrix operand order repaired and checked by reading, not compiled, not run on hardware
//
#include <hip/hip_runtime.h>
#include <hip/hip_bf16.h>

typedef __attribute__((ext_vector_type(16))) _Float16 v16h;
typedef __attribute__((ext_vector_type(8)))  float    v8f;
typedef __attribute__((ext_vector_type(8)))  unsigned int v8u;

#define NN 50000
#define MP 50048          // NN padded to a multiple of 64 (782 * 64)
#define EE 800000
#define FIN 128
#define EDIM 16
#define EMBC 32
#define NH 4
#define HCH 128
#define NB 64
#define NEG_INF (-__builtin_inff())

// ---------------- utility device helpers ----------------

__device__ __forceinline__ float waveReduceSum(float v) {
#pragma unroll
  for (int off = 16; off > 0; off >>= 1) v += __shfl_xor(v, off, 32);
  return v;
}

__device__ __forceinline__ void atomicMaxF(float* addr, float val) {
  unsigned int* a = (unsigned int*)addr;
  unsigned int old = *a;
  while (__uint_as_float(old) < val) {
    unsigned int assumed = old;
    old = atomicCAS(a, assumed, __float_as_uint(val));
    if (old == assumed) break;
  }
}

__global__ void k_fill(float* __restrict__ p, long n, float v) {
  long i = (long)blockIdx.x * blockDim.x + threadIdx.x;
  long stride = (long)gridDim.x * blockDim.x;
  for (; i < n; i += stride) p[i] = v;
}

__global__ void k_copy_pad(const float* __restrict__ src, float* __restrict__ dst,
                           int nsrc, int ntot) {
  int i = blockIdx.x * blockDim.x + threadIdx.x;
  if (i < ntot) dst[i] = (i < nsrc) ? src[i] : 0.0f;
}

// single v_cvt_pk_rtz_f16_f32: correct pairing, no half re-swizzle
__device__ __forceinline__ unsigned int packpair(float f0, float f1) {
  auto p = __builtin_amdgcn_cvt_pkrtz(f0, f1);  // __fp16 ext_vector(2)
  return __builtin_bit_cast(unsigned int, p);
}

// ---------------- weight packing into WMMA B-fragment layout ----------------
// B fragment (16-bit, 32-K tile x 16 cols): lane l holds column n = l&15,
// kb = (l<16)?0:16, dword j holds K = kb+2j, kb+2j+1.
__global__ void k_pack_wfrag(const float* __restrict__ W, unsigned int* __restrict__ frag,
                             int K, int Nout) {
  int ctiles = Nout >> 4;
  int ktiles = (K + 31) >> 5;
  int total = ktiles * ctiles * 32 * 8;
  int idx = blockIdx.x * blockDim.x + threadIdx.x;
  if (idx >= total) return;
  int j    = idx & 7;
  int lane = (idx >> 3) & 31;
  int tile = idx >> 8;
  int c = tile % ctiles;
  int t = tile / ctiles;
  int n  = lane & 15;
  int kb = (lane < 16) ? 0 : 16;
  int k0 = t * 32 + kb + 2 * j;
  float f0 = (k0     < K) ? W[k0 * Nout + c * 16 + n] : 0.0f;
  float f1 = (k0 + 1 < K) ? W[(k0 + 1) * Nout + c * 16 + n] : 0.0f;
  frag[idx] = packpair(f0, f1);
}

// ---------------- WMMA GEMM core (64 rows/wave via 4 accumulators) ----------------
// A has MP rows (padded, multiple of 64): no guards anywhere in the hot loop.
// KK and CTB are compile-time so everything unrolls into straight-line code.
// Staging is split into a load phase (all b128 loads in flight as a clause) and
// a convert/store phase (v_cvt_pk_rtz_f16_f32 pairs -> one ds_store_b64 each).

#define GEMM_CORE(KK, CTB, FRAG_EXPR, EPILOGUE)                                     \
  __shared__ unsigned int As[64 * 18];                                              \
  const int lane = threadIdx.x;                                                     \
  const int wv   = threadIdx.y;                                                     \
  const int tid  = wv * 32 + lane;                                                  \
  const int row0 = blockIdx.x * 64;                                                 \
  constexpr int ksteps = KK >> 5;                                                   \
  constexpr int nth    = CTB * 32;                                                  \
  constexpr int nq     = 512 / nth;                                                 \
  v8f acc[4] = {};                                                                  \
  const int m   = lane & 15;                                                        \
  const int kb8 = (lane < 16) ? 0 : 16;                                             \
  _Pragma("unroll") for (int kc = 0; kc < ksteps; ++kc) {                           \
    float4 fbuf[nq];                                                                \
    _Pragma("unroll") for (int s = 0; s < nq; ++s) {                                \
      int i  = tid + s * nth;   /* i indexes 512 float4 quads of the 64x32 chunk */ \
      int mm = i >> 3;                                                              \
      int kq = i & 7;                                                               \
      const float4* ap = (const float4*)(A + (row0 + mm) * KK + kc * 32 + kq * 4);  \
      fbuf[s] = *ap;                                                                \
      if (kc + 1 < ksteps) __builtin_prefetch((const char*)ap + 128, 0, 3);         \
    }                                                                               \
    _Pragma("unroll") for (int s = 0; s < nq; ++s) {                                \
      int i  = tid + s * nth;                                                       \
      int mm = i >> 3;                                                              \
      int kq = i & 7;                                                               \
      uint2 u;                                                                      \
      u.x = packpair(fbuf[s].x, fbuf[s].y);                                         \
      u.y = packpair(fbuf[s].z, fbuf[s].w);                                         \
      *(uint2*)&As[mm * 18 + kq * 2] = u;                                           \
    }                                                                               \
    __syncthreads();                                                                \
    const unsigned int* bp = frag + (FRAG_EXPR) * 256 + lane * 8;                   \
    v8u braw;                                                                       \
    _Pragma("unroll") for (int jj = 0; jj < 8; ++jj) braw[jj] = bp[jj];             \
    v16h b = __builtin_bit_cast(v16h, braw);                                        \
    _Pragma("unroll") for (int rt = 0; rt < 4; ++rt) {                              \
      v16h a;                                                                       \
      _Pragma("unroll") for (int jj = 0; jj < 8; ++jj) {                            \
        int kk = kb8 + 2 * jj;                                                      \
        unsigned int u = As[(rt * 16 + m) * 18 + (kk >> 1)];                        \
        a[2 * jj]     = __builtin_bit_cast(_Float16, (unsigned short)(u & 0xffffu)); \
        a[2 * jj + 1] = __builtin_bit_cast(_Float16, (unsigned short)(u >> 16));    \
      }                                                                             \
      acc[rt] = __builtin_amdgcn_wmma_f32_16x16x32_f16(false, a, false, b,          \
                                                       (short)0, acc[rt], false, false); \
    }                                                                               \
    __syncthreads();                                                                \
  }                                                                                 \
  const int n  = lane & 15;                                                         \
  const int mb = (lane < 16) ? 0 : 8;                                               \
  EPILOGUE

// Fused q/k/v/skip GEMM: 4 matrices x Nout=128, single pass over A.
struct Outs4 { float* o0; float* o1; float* o2; float* o3;
               const float* b0; const float* b1; const float* b2; const float* b3; };

template <int KK>
__global__ void __launch_bounds__(128)
k_gemm_qkvs(const float* __restrict__ A, const unsigned int* __restrict__ frag,
            Outs4 ob) {
  const int ctg  = blockIdx.y * 4 + threadIdx.y;  // 0..31
  const int mat  = ctg >> 3;
  const int ctl  = ctg & 7;
  const int col0 = ctl * 16;
  GEMM_CORE(KK, 4, ((mat * (KK >> 5) + kc) * 8 + ctl),
    {
      float* C = (mat == 0) ? ob.o0 : (mat == 1) ? ob.o1 : (mat == 2) ? ob.o2 : ob.o3;
      const float* bias = (mat == 0) ? ob.b0 : (mat == 1) ? ob.b1 : (mat == 2) ? ob.b2 : ob.b3;
      float bv = bias[col0 + n];
      _Pragma("unroll") for (int rt = 0; rt < 4; ++rt) {
        _Pragma("unroll") for (int r = 0; r < 8; ++r)
          C[(row0 + rt * 16 + mb + r) * HCH + col0 + n] = acc[rt][r] + bv;
      }
    })
}

// transf GEMM: Nout=32 (2 col tiles), bias + ReLU epilogue.
template <int KK>
__global__ void __launch_bounds__(64)
k_gemm_transf(const float* __restrict__ A, const unsigned int* __restrict__ frag,
              const float* __restrict__ bias, float* __restrict__ C) {
  const int ct   = threadIdx.y;  // 0..1
  const int col0 = ct * 16;
  GEMM_CORE(KK, 2, (kc * 2 + ct),
    {
      float bv = bias[col0 + n];
      _Pragma("unroll") for (int rt = 0; rt < 4; ++rt) {
        _Pragma("unroll") for (int r = 0; r < 8; ++r) {
          float v = acc[rt][r] + bv;
          v = v > 0.0f ? v : 0.0f;
          C[(row0 + rt * 16 + mb + r) * EMBC + col0 + n] = v;
        }
      }
    })
}

// ---------------- edge phase: alpha = (q[dst] . (k[src]+e)) / sqrt(EMB) -----------
__global__ void __launch_bounds__(256)
k_edge_alpha(const float* __restrict__ q, const float* __restrict__ kk,
             const float* __restrict__ ea, const float* __restrict__ We,
             const int* __restrict__ ei, float* __restrict__ alpha,
             float* __restrict__ amax) {
  __shared__ float Ws[EDIM * HCH];
  for (int i = threadIdx.x; i < EDIM * HCH; i += 256) Ws[i] = We[i];
  __syncthreads();
  const int lane = threadIdx.x & 31;
  const int wv   = threadIdx.x >> 5;
  int e = blockIdx.x * 8 + wv;
  if (e >= EE) return;
  const int src = ei[e];
  const int dst = ei[EE + e];
  float eav[EDIM];
#pragma unroll
  for (int i = 0; i < EDIM; ++i) eav[i] = ea[e * EDIM + i];
  const float scale = 0.17677669529663687f;  // 1/sqrt(32)
#pragma unroll
  for (int h = 0; h < NH; ++h) {
    int c = h * EMBC + lane;
    float ev = 0.0f;
#pragma unroll
    for (int i = 0; i < EDIM; ++i) ev += eav[i] * Ws[i * HCH + c];
    float kj = kk[src * HCH + c] + ev;
    float qd = q[dst * HCH + c];
    float s = waveReduceSum(kj * qd) * scale;
    if (lane == 0) {
      alpha[e * NH + h] = s;
      atomicMaxF(&amax[dst * NH + h], s);
    }
  }
}

__global__ void k_edge_exp(float* __restrict__ alpha, const int* __restrict__ ei,
                           const float* __restrict__ amax, float* __restrict__ den) {
  int i = blockIdx.x * blockDim.x + threadIdx.x;
  if (i >= EE * NH) return;
  int e = i >> 2;
  int h = i & 3;
  int dst = ei[EE + e];
  float am = amax[dst * NH + h];
  if (!(am > -1e37f)) am = 0.0f;  // mirror isfinite guard
  float ex = __expf(alpha[i] - am);
  alpha[i] = ex;
  atomicAdd(&den[dst * NH + h], ex);
}

__global__ void __launch_bounds__(256)
k_edge_msg(const float* __restrict__ v, const float* __restrict__ ea,
           const float* __restrict__ We, const int* __restrict__ ei,
           const float* __restrict__ ex, const float* __restrict__ den,
           float* __restrict__ outb) {
  __shared__ float Ws[EDIM * HCH];
  for (int i = threadIdx.x; i < EDIM * HCH; i += 256) Ws[i] = We[i];
  __syncthreads();
  const int lane = threadIdx.x & 31;
  const int wv   = threadIdx.x >> 5;
  int e = blockIdx.x * 8 + wv;
  if (e >= EE) return;
  const int src = ei[e];
  const int dst = ei[EE + e];
  float eav[EDIM];
#pragma unroll
  for (int i = 0; i < EDIM; ++i) eav[i] = ea[e * EDIM + i];
#pragma unroll
  for (int h = 0; h < NH; ++h) {
    float w = ex[e * NH + h] / (den[dst * NH + h] + 1e-16f);
    int c = h * EMBC + lane;
    float ev = 0.0f;
#pragma unroll
    for (int i = 0; i < EDIM; ++i) ev += eav[i] * Ws[i * HCH + c];
    float msg = (v[src * HCH + c] + ev) * w;
    atomicAdd(&outb[dst * HCH + c], msg);
  }
}

// ---------------- beta gate + combine ----------------
__global__ void __launch_bounds__(256)
k_combine(const float* __restrict__ outb, const float* __restrict__ skip,
          const float* __restrict__ Wb, float* __restrict__ hcomb) {
  const int lane = threadIdx.x & 31;
  const int wv   = threadIdx.x >> 5;
  int nd = blockIdx.x * 8 + wv;
  if (nd >= NN) return;
  float o[4], sk[4], s = 0.0f;
#pragma unroll
  for (int j = 0; j < 4; ++j) {
    int c = j * 32 + lane;
    o[j]  = outb[nd * HCH + c];
    sk[j] = skip[nd * HCH + c];
    s += o[j] * Wb[c] + sk[j] * Wb[HCH + c] + (o[j] - sk[j]) * Wb[2 * HCH + c];
  }
  s = waveReduceSum(s);
  float beta = 1.0f / (1.0f + __expf(-s));
#pragma unroll
  for (int j = 0; j < 4; ++j) {
    int c = j * 32 + lane;
    hcomb[nd * HCH + c] = beta * sk[j] + (1.0f - beta) * o[j];
  }
}

// ---------------- batchnorm ----------------
__global__ void k_bn_stats(const float* __restrict__ h, float* __restrict__ st) {
  int total = NN * EMBC;
  int stride = gridDim.x * blockDim.x;  // multiple of 32
  int i = blockIdx.x * blockDim.x + threadIdx.x;
  int c = i & 31;
  float s = 0.f, ss = 0.f;
  for (; i < total; i += stride) { float v = h[i]; s += v; ss += v * v; }
  atomicAdd(&st[c], s);
  atomicAdd(&st[32 + c], ss);
}

__global__ void k_bn_apply(float* __restrict__ h, const float* __restrict__ st,
                           const float* __restrict__ g, const float* __restrict__ b) {
  int i = blockIdx.x * blockDim.x + threadIdx.x;
  if (i >= NN * EMBC) return;
  int c = i & 31;
  float mu  = st[c] * (1.0f / NN);
  float var = st[32 + c] * (1.0f / NN) - mu * mu;
  h[i] = (h[i] - mu) * rsqrtf(var + 1e-5f) * g[c] + b[c];
}

// ---------------- global pool (max + mean per batch) ----------------
__global__ void k_pool_acc(const float* __restrict__ h, const int* __restrict__ bidx,
                           float* __restrict__ pmax, float* __restrict__ psum,
                           float* __restrict__ cnt) {
  int i = blockIdx.x * blockDim.x + threadIdx.x;
  if (i >= NN * EMBC) return;
  int nd = i >> 5;
  int c = i & 31;
  int b = bidx[nd];
  float v = h[i];
  atomicMaxF(&pmax[b * EMBC + c], v);
  atomicAdd(&psum[b * EMBC + c], v);
  if (c == 0) atomicAdd(&cnt[b], 1.0f);
}

__global__ void k_pool_final(const float* __restrict__ pmax, const float* __restrict__ psum,
                             const float* __restrict__ cnt, float* __restrict__ outp) {
  int i = blockIdx.x * blockDim.x + threadIdx.x;
  if (i >= NB * EMBC) return;
  int b = i >> 5;
  int c = i & 31;
  float ct = cnt[b];
  float mx = pmax[b * EMBC + c];
  if (!(ct > 0.0f)) mx = 0.0f;
  float mean = psum[b * EMBC + c] / fmaxf(ct, 1.0f);
  outp[b * 2 * EMBC + c]        += mx;
  outp[b * 2 * EMBC + EMBC + c] += mean;
}

// ---------------- host orchestration ----------------
extern "C" void kernel_launch(void* const* d_in, const int* in_sizes, int n_in,
                              void* d_out, int out_size, void* d_ws, size_t ws_size,
                              hipStream_t stream) {
  const float* x         = (const float*)d_in[0];
  const float* edge_attr = (const float*)d_in[1];
  const int*   ei        = (const int*)d_in[2];
  const int*   bidx      = (const int*)d_in[3];

  struct ConvP { const float *qW,*qb,*kW,*kb,*vW,*vb,*eW,*skW,*skb,*betaW; };
  auto getConv = [&](int base) {
    ConvP c;
    c.qW  = (const float*)d_in[base + 0]; c.qb  = (const float*)d_in[base + 1];
    c.kW  = (const float*)d_in[base + 2]; c.kb  = (const float*)d_in[base + 3];
    c.vW  = (const float*)d_in[base + 4]; c.vb  = (const float*)d_in[base + 5];
    c.eW  = (const float*)d_in[base + 6];
    c.skW = (const float*)d_in[base + 7]; c.skb = (const float*)d_in[base + 8];
    c.betaW = (const float*)d_in[base + 9];
    return c;
  };
  ConvP conv1 = getConv(4);
  const float* t1W  = (const float*)d_in[14]; const float* t1b  = (const float*)d_in[15];
  const float* bn1g = (const float*)d_in[16]; const float* bn1b = (const float*)d_in[17];
  ConvP convs[2] = { getConv(18), getConv(28) };
  const float* tW[2]  = {(const float*)d_in[38], (const float*)d_in[40]};
  const float* tb[2]  = {(const float*)d_in[39], (const float*)d_in[41]};
  const float* bng[2] = {(const float*)d_in[42], (const float*)d_in[44]};
  const float* bnb[2] = {(const float*)d_in[43], (const float*)d_in[45]};

  // workspace carve (floats) -- all GEMM activations padded to MP rows
  float* ws = (float*)d_ws;
  long o = 0;
  float* xpad  = ws + o; o += (long)MP * FIN;
  float* qbuf  = ws + o; o += (long)MP * HCH;
  float* kbuf  = ws + o; o += (long)MP * HCH;
  float* vbuf  = ws + o; o += (long)MP * HCH;
  float* sbuf  = ws + o; o += (long)MP * HCH;
  float* outb  = ws + o; o += (long)MP * HCH;
  float* hcomb = ws + o; o += (long)MP * HCH;
  float* hemb  = ws + o; o += (long)MP * EMBC;
  float* alpha = ws + o; o += (long)EE * NH;
  float* amax  = ws + o; o += (long)NN * NH;
  float* den   = ws + o; o += (long)NN * NH;
  float* st    = ws + o; o += 64;
  float* pmax  = ws + o; o += NB * EMBC;
  float* psum  = ws + o; o += NB * EMBC;
  float* cnt   = ws + o; o += NB;
  unsigned int* fqkvs = (unsigned int*)(ws + o); o += 4 * 8192;  // 4 mats, max din=128
  unsigned int* ft    = (unsigned int*)(ws + o); o += 2048;

  auto fill = [&](float* p, long n, float v) {
    int blocks = (int)((n + 255) / 256); if (blocks > 1024) blocks = 1024;
    k_fill<<<blocks, 256, 0, stream>>>(p, n, v);
  };
  auto pack = [&](const float* W, unsigned int* f, int K, int Nout) {
    int total = ((K + 31) / 32) * (Nout / 16) * 256;
    k_pack_wfrag<<<(total + 255) / 256, 256, 0, stream>>>(W, f, K, Nout);
  };

  const int rblocks = MP / 64;  // 782 row blocks, exact

  // pad x once; zero hcomb's pad rows (never written by combine, read by transf GEMM)
  k_copy_pad<<<(MP * FIN + 255) / 256, 256, 0, stream>>>(x, xpad, NN * FIN, MP * FIN);
  fill(hcomb + (long)NN * HCH, (long)(MP - NN) * HCH, 0.0f);

  auto conv = [&](const float* hin, int din, const ConvP& p) {
    int fragsz = (din / 32) * 8 * 256;  // dwords per matrix
    pack(p.qW,  fqkvs,              din, HCH);
    pack(p.kW,  fqkvs + fragsz,     din, HCH);
    pack(p.vW,  fqkvs + 2 * fragsz, din, HCH);
    pack(p.skW, fqkvs + 3 * fragsz, din, HCH);
    Outs4 ob = { qbuf, kbuf, vbuf, sbuf, p.qb, p.kb, p.vb, p.skb };
    if (din == FIN)
      k_gemm_qkvs<FIN><<<dim3(rblocks, 8), dim3(32, 4), 0, stream>>>(hin, fqkvs, ob);
    else
      k_gemm_qkvs<EMBC><<<dim3(rblocks, 8), dim3(32, 4), 0, stream>>>(hin, fqkvs, ob);
    fill(amax, (long)NN * NH, NEG_INF);
    fill(den,  (long)NN * NH, 0.0f);
    fill(outb, (long)NN * HCH, 0.0f);
    int eblocks = (EE + 7) / 8;
    k_edge_alpha<<<eblocks, 256, 0, stream>>>(qbuf, kbuf, edge_attr, p.eW, ei, alpha, amax);
    k_edge_exp<<<(EE * NH + 255) / 256, 256, 0, stream>>>(alpha, ei, amax, den);
    k_edge_msg<<<eblocks, 256, 0, stream>>>(vbuf, edge_attr, p.eW, ei, alpha, den, outb);
    k_combine<<<(NN + 7) / 8, 256, 0, stream>>>(outb, sbuf, p.betaW, hcomb);
  };

  auto transf_bn = [&](const float* W, const float* b, const float* g, const float* bb) {
    pack(W, ft, HCH, EMBC);
    k_gemm_transf<HCH><<<dim3(rblocks, 1), dim3(32, 2), 0, stream>>>(hcomb, ft, b, hemb);
    fill(st, 64, 0.0f);
    k_bn_stats<<<512, 256, 0, stream>>>(hemb, st);
    k_bn_apply<<<(NN * EMBC + 255) / 256, 256, 0, stream>>>(hemb, st, g, bb);
  };

  // layer 1
  conv(xpad, FIN, conv1);
  transf_bn(t1W, t1b, bn1g, bn1b);

  // loop layers with pooling accumulation
  fill((float*)d_out, NB * 2 * EMBC, 0.0f);
  for (int i = 0; i < 2; ++i) {
    conv(hemb, EMBC, convs[i]);
    transf_bn(tW[i], tb[i], bng[i], bnb[i]);
    fill(pmax, NB * EMBC, NEG_INF);
    fill(psum, NB * EMBC, 0.0f);
    fill(cnt,  NB, 0.0f);
    k_pool_acc<<<(NN * EMBC + 255) / 256, 256, 0, stream>>>(hemb, bidx, pmax, psum, cnt);
    k_pool_final<<<(NB * EMBC + 255) / 256, 256, 0, stream>>>(pmax, psum, cnt, (float*)d_out);
  }
}
